// AttnBlock_73091753443778
// MI455X (gfx1250) — matrix operand order repaired and checked by reading, not compiled, not run on hardware
//
#include <hip/hip_runtime.h>

// ---------------------------------------------------------------------------
// AttnBlock for MI455X (gfx1250, wave32, WMMA bf16 16x16x32, f32 accumulate)
// B=8, C=256, HW=4096, d=128. Flash-attention streaming softmax, fully fused
// output projection + residual. All GEMMs via v_wmma_f32_16x16x32_bf16.
// ---------------------------------------------------------------------------

typedef __attribute__((ext_vector_type(16))) __bf16 v16bf;
typedef __attribute__((ext_vector_type(8)))  __bf16 v8bf;
typedef __attribute__((ext_vector_type(8)))  float  v8f;

#define HW_   4096
#define CCH   256
#define DDIM  128
#define NBAT  8
#define QKSCALE 0.08838834764831845f   // 128^-0.5

// ---- WMMA helper -----------------------------------------------------------
static __device__ __forceinline__ v8f wmma_bf16(v16bf a, v16bf b, v8f c) {
  return __builtin_amdgcn_wmma_f32_16x16x32_bf16(false, a, false, b,
                                                 (short)0, c, false, false);
}

// A fragment (16x32 bf16, M x K), source row-major with leading dim ld.
// Lane L<16: M=L, elems 0..7 = K 0..7, elems 8..15 = K 16..23.
// Lane L>=16: same M, K shifted by 8.
static __device__ __forceinline__ v16bf load_afrag(const __bf16* base, int ld) {
  int lane = threadIdx.x & 31;
  int lh = lane >> 4, r = lane & 15;
  const __bf16* p = base + (size_t)r * ld + 8 * lh;
  v16bf a;
  *((v8bf*)&a)       = *(const v8bf*)(p);
  *(((v8bf*)&a) + 1) = *(const v8bf*)(p + 16);
  return a;
}

// B fragment (32x16 bf16, K x N), source = row-major B^T (rows = N) with ld.
// Lane n<16: col n, elems 0..7 = K 0..7, elems 8..15 = K 16..23. Lane n>=16: K shifted by 8.
static __device__ __forceinline__ v16bf load_bfrag(const __bf16* base, int ld) {
  int lane = threadIdx.x & 31;
  int lh = lane >> 4, n = lane & 15;
  const __bf16* p = base + (size_t)n * ld + 8 * lh;
  v16bf b;
  *((v8bf*)&b)       = *(const v8bf*)(p);
  *(((v8bf*)&b) + 1) = *(const v8bf*)(p + 16);
  return b;
}

// ---- Kernel 1: [C][HW] f32 -> [HW][C] bf16 tiled transpose -----------------
__global__ void k_transpose(const float* __restrict__ x, const float* __restrict__ y,
                            __bf16* __restrict__ xbf, __bf16* __restrict__ ybf) {
  __shared__ float tile[32][33];
  int which = blockIdx.z & 1, b = blockIdx.z >> 1;
  const float* src = which ? y : x;
  __bf16* dst = which ? ybf : xbf;
  int n0 = blockIdx.x * 32, c0 = blockIdx.y * 32;
  int tx = threadIdx.x, ty = threadIdx.y;          // block (32, 8)
  const float* s = src + (size_t)b * CCH * HW_;
  #pragma unroll
  for (int i = 0; i < 4; ++i)
    tile[ty + i * 8][tx] = s[(size_t)(c0 + ty + i * 8) * HW_ + n0 + tx];
  __syncthreads();
  __bf16* d = dst + (size_t)b * HW_ * CCH;
  #pragma unroll
  for (int i = 0; i < 4; ++i)
    d[(size_t)(n0 + ty + i * 8) * CCH + c0 + tx] = (__bf16)tile[tx][ty + i * 8];
}

// ---- Kernel 2: weights -> bf16 (Wq scaled by 1/sqrt(d)) --------------------
__global__ void k_weights(const float* __restrict__ Wq, const float* __restrict__ Wk,
                          const float* __restrict__ Wv, const float* __restrict__ Wo,
                          __bf16* __restrict__ Wqb, __bf16* __restrict__ Wkb,
                          __bf16* __restrict__ Wvb, __bf16* __restrict__ Wob) {
  int i = blockIdx.x * blockDim.x + threadIdx.x;
  if (i < 32768)        Wqb[i]           = (__bf16)(Wq[i] * QKSCALE);
  else if (i < 65536)   Wkb[i - 32768]   = (__bf16)Wk[i - 32768];
  else if (i < 131072)  Wvb[i - 65536]   = (__bf16)Wv[i - 65536];
  else if (i < 196608)  Wob[i - 131072]  = (__bf16)Wo[i - 131072];
}

// ---- Kernel 3: projections Q, K (row-major), V^T ---------------------------
// grid (HW/16, B), block 256 (8 waves, 4 jobs each of 32 total).
__global__ __launch_bounds__(256) void k_proj(
    const __bf16* __restrict__ xbf, const __bf16* __restrict__ ybf,
    const __bf16* __restrict__ Wqb, const __bf16* __restrict__ Wkb,
    const __bf16* __restrict__ Wvb,
    const float* __restrict__ bq, const float* __restrict__ bk,
    const float* __restrict__ bv,
    __bf16* __restrict__ Qb, __bf16* __restrict__ Kb, __bf16* __restrict__ Vt) {
  int b  = blockIdx.y;
  int n0 = blockIdx.x * 16;
  int wid  = threadIdx.x >> 5;
  int lane = threadIdx.x & 31, lh = lane >> 4, r_ = lane & 15;
  const __bf16* xb_ = xbf + (size_t)b * HW_ * CCH;
  const __bf16* yb_ = ybf + (size_t)b * HW_ * CCH;

  for (int jj = 0; jj < 4; ++jj) {
    int j = wid * 4 + jj;
    v8f acc; 
    #pragma unroll
    for (int i = 0; i < 8; ++i) acc[i] = 0.f;

    if (j < 16) {                         // Q (j<8) or K (j<16) d-tile
      int dt = j & 7;
      const __bf16* src = (j < 8) ? yb_ : xb_;
      const __bf16* W   = (j < 8) ? Wqb : Wkb;
      #pragma unroll
      for (int c = 0; c < 8; ++c) {
        v16bf a  = load_afrag(src + (size_t)n0 * CCH + c * 32, CCH);
        v16bf bb = load_bfrag(W + (size_t)(dt * 16) * CCH + c * 32, CCH);
        acc = wmma_bf16(a, bb, acc);
      }
      const float* bias = (j < 8) ? bq : bk;
      float bsc = (j < 8) ? QKSCALE : 1.0f;
      __bf16* out = ((j < 8) ? Qb : Kb) + (size_t)b * HW_ * DDIM;
      int col = dt * 16 + r_;
      float bval = bias[col] * bsc;
      #pragma unroll
      for (int r = 0; r < 8; ++r) {
        int row = n0 + r + 8 * lh;
        out[(size_t)row * DDIM + col] = (__bf16)(acc[r] + bval);
      }
    } else {                              // V^T e-tile: rows e, cols n
      int et = j - 16;
      #pragma unroll
      for (int c = 0; c < 8; ++c) {
        v16bf a  = load_afrag(Wvb + (size_t)(et * 16) * CCH + c * 32, CCH);
        v16bf bb = load_bfrag(xb_ + (size_t)n0 * CCH + c * 32, CCH);
        acc = wmma_bf16(a, bb, acc);
      }
      __bf16* out = Vt + (size_t)b * CCH * HW_;
      int n = n0 + r_;
      #pragma unroll
      for (int r = 0; r < 8; ++r) {
        int e = et * 16 + r + 8 * lh;
        out[(size_t)e * HW_ + n] = (__bf16)(acc[r] + bv[e]);
      }
    }
  }
}

// ---- Kernel 4: flash attention + fused output projection + residual --------
// grid (HW/32, B), block 128 = 4 waves = 2 pairs. Each pair: 16 query rows;
// both waves of a pair compute S/softmax redundantly, split e (128 each).
__global__ __launch_bounds__(128) void k_attn(
    const __bf16* __restrict__ Qb, const __bf16* __restrict__ Kb,
    const __bf16* __restrict__ Vt, const __bf16* __restrict__ Wob,
    const float* __restrict__ bo, const float* __restrict__ x,
    float* __restrict__ out) {
  __shared__ __attribute__((aligned(16))) __bf16 Plds[4][16 * 32];
  __shared__ __attribute__((aligned(16))) __bf16 Hlds[2][16 * 256];

  int b = blockIdx.y;
  int wid  = threadIdx.x >> 5;
  int pair = wid >> 1, ePart = wid & 1;
  int n0 = blockIdx.x * 32 + pair * 16;
  int lane = threadIdx.x & 31, lh = lane >> 4, r_ = lane & 15;

  const __bf16* Q = Qb + (size_t)b * HW_ * DDIM;
  const __bf16* K = Kb + (size_t)b * HW_ * DDIM;
  const __bf16* V = Vt + (size_t)b * CCH * HW_;

  v16bf qf[4];
  #pragma unroll
  for (int c = 0; c < 4; ++c) qf[c] = load_afrag(Q + (size_t)n0 * DDIM + c * 32, DDIM);

  v8f O[8];
  #pragma unroll
  for (int t = 0; t < 8; ++t)
    #pragma unroll
    for (int i = 0; i < 8; ++i) O[t][i] = 0.f;
  float m_i[8], l_i[8];
  #pragma unroll
  for (int r = 0; r < 8; ++r) { m_i[r] = -1e30f; l_i[r] = 0.f; }

  __bf16* Pb = &Plds[wid][0];

  for (int kb = 0; kb < 128; ++kb) {
    int k0 = kb * 32;
    v8f S0, S1;
    #pragma unroll
    for (int i = 0; i < 8; ++i) { S0[i] = 0.f; S1[i] = 0.f; }
    #pragma unroll
    for (int c = 0; c < 4; ++c) {                 // S = Q K^T (scale in Wq)
      v16bf kb0 = load_bfrag(K + (size_t)k0 * DDIM + c * 32, DDIM);
      v16bf kb1 = load_bfrag(K + (size_t)(k0 + 16) * DDIM + c * 32, DDIM);
      S0 = wmma_bf16(qf[c], kb0, S0);
      S1 = wmma_bf16(qf[c], kb1, S1);
    }
    // online softmax: VGPR r holds row (r + 8*lh); cols live in 16-lane group
    float corr[8]; float changed = 0.f;
    #pragma unroll
    for (int r = 0; r < 8; ++r) {
      float mx = fmaxf(S0[r], S1[r]);
      mx = fmaxf(mx, __shfl_xor(mx, 8));
      mx = fmaxf(mx, __shfl_xor(mx, 4));
      mx = fmaxf(mx, __shfl_xor(mx, 2));
      mx = fmaxf(mx, __shfl_xor(mx, 1));
      float mnew = fmaxf(m_i[r], mx);
      if (mnew != m_i[r]) changed = 1.f;
      corr[r] = __expf(m_i[r] - mnew);
      float p0 = __expf(S0[r] - mnew);
      float p1 = __expf(S1[r] - mnew);
      float rs = p0 + p1;
      rs += __shfl_xor(rs, 8);
      rs += __shfl_xor(rs, 4);
      rs += __shfl_xor(rs, 2);
      rs += __shfl_xor(rs, 1);
      l_i[r] = l_i[r] * corr[r] + rs;
      m_i[r] = mnew;
      S0[r] = p0; S1[r] = p1;
    }
    // stage P (D layout -> row-major LDS -> A fragment)
    asm volatile("" ::: "memory");
    #pragma unroll
    for (int r = 0; r < 8; ++r) {
      Pb[(r + 8 * lh) * 32 + r_]      = (__bf16)S0[r];
      Pb[(r + 8 * lh) * 32 + 16 + r_] = (__bf16)S1[r];
    }
    asm volatile("" ::: "memory");
    v16bf pf = load_afrag(Pb, 32);
    if (__any(changed != 0.f)) {
      #pragma unroll
      for (int t = 0; t < 8; ++t)
        #pragma unroll
        for (int r = 0; r < 8; ++r) O[t][r] *= corr[r];
    }
    #pragma unroll
    for (int t = 0; t < 8; ++t) {                 // O += P * V (this e-half)
      v16bf vb = load_bfrag(V + (size_t)(ePart * 128 + t * 16) * HW_ + k0, HW_);
      O[t] = wmma_bf16(pf, vb, O[t]);
    }
  }

  // normalize, publish h (16 x 256 bf16) to pair-shared LDS
  __bf16* Hb = &Hlds[pair][0];
  #pragma unroll
  for (int r = 0; r < 8; ++r) {
    float inv = 1.0f / l_i[r];
    int row = r + 8 * lh;
    #pragma unroll
    for (int t = 0; t < 8; ++t)
      Hb[row * 256 + ePart * 128 + t * 16 + r_] = (__bf16)(O[t][r] * inv);
  }
  __syncthreads();

  // fused out-proj: out[f][n] = x[f][n] + sum_e Wo[f][e] h[n][e] + bo[f]
  const float* xg = x + (size_t)b * CCH * HW_;
  float* og = out + (size_t)b * CCH * HW_;
  for (int ft = 0; ft < 8; ++ft) {
    int f0 = ePart * 128 + ft * 16;
    v8f acc;
    #pragma unroll
    for (int i = 0; i < 8; ++i) acc[i] = 0.f;
    #pragma unroll
    for (int c = 0; c < 8; ++c) {
      v16bf a  = load_afrag(Wob + (size_t)f0 * CCH + c * 32, CCH);
      v16bf bb = load_bfrag(Hb + c * 32, 256);
      acc = wmma_bf16(a, bb, acc);
    }
    int n = n0 + r_;
    #pragma unroll
    for (int r = 0; r < 8; ++r) {
      int f = f0 + r + 8 * lh;
      og[(size_t)f * HW_ + n] = xg[(size_t)f * HW_ + n] + acc[r] + bo[f];
    }
  }
}

// ---------------------------------------------------------------------------
extern "C" void kernel_launch(void* const* d_in, const int* in_sizes, int n_in,
                              void* d_out, int out_size, void* d_ws, size_t ws_size,
                              hipStream_t stream) {
  (void)in_sizes; (void)n_in; (void)out_size; (void)ws_size;
  const float* x  = (const float*)d_in[0];
  const float* y  = (const float*)d_in[1];
  const float* Wq = (const float*)d_in[2];
  const float* bq = (const float*)d_in[3];
  const float* Wk = (const float*)d_in[4];
  const float* bk = (const float*)d_in[5];
  const float* Wv = (const float*)d_in[6];
  const float* bv = (const float*)d_in[7];
  const float* Wo = (const float*)d_in[8];
  const float* bo = (const float*)d_in[9];
  float* out = (float*)d_out;

  char* ws = (char*)d_ws;                 // needs ~67.6 MB workspace
  __bf16* xbf = (__bf16*)(ws);                       // 16 MB
  __bf16* ybf = (__bf16*)(ws + 16777216);            // 16 MB
  __bf16* Qb  = (__bf16*)(ws + 33554432);            // 8 MB
  __bf16* Kb  = (__bf16*)(ws + 41943040);            // 8 MB
  __bf16* Vt  = (__bf16*)(ws + 50331648);            // 16 MB
  __bf16* Wqb = (__bf16*)(ws + 67108864);
  __bf16* Wkb = (__bf16*)(ws + 67174400);
  __bf16* Wvb = (__bf16*)(ws + 67239936);
  __bf16* Wob = (__bf16*)(ws + 67371008);

  k_transpose<<<dim3(HW_ / 32, CCH / 32, NBAT * 2), dim3(32, 8), 0, stream>>>(
      x, y, xbf, ybf);
  k_weights<<<dim3(768), dim3(256), 0, stream>>>(Wq, Wk, Wv, Wo, Wqb, Wkb, Wvb, Wob);
  k_proj<<<dim3(HW_ / 16, NBAT), dim3(256), 0, stream>>>(
      xbf, ybf, Wqb, Wkb, Wvb, bq, bk, bv, Qb, Kb, Vt);
  k_attn<<<dim3(HW_ / 32, NBAT), dim3(128), 0, stream>>>(
      Qb, Kb, Vt, Wob, bo, x, out);
}
